// GIN_33861522162133
// MI455X (gfx1250) — matrix operand order repaired and checked by reading, not compiled, not run on hardware
//
#include <hip/hip_runtime.h>
#include <hip/hip_bf16.h>

// GIN forward for MI455X (gfx1250, wave32).
//  - aggregation: f32 global atomics (node table + agg buffer are L2-resident: 51MB each vs 192MB L2)
//  - GEMMs: v_wmma_f32_16x16x32_f16, f32 accumulate; weights staged in LDS as f16 (swizzled banks)
//  - BN (training-mode batch stats): two-pass column sums, BN+ReLU fused into next GEMM's A load.
//  - K and M are compile-time so store addressing folds into immediate offsets.

typedef __attribute__((ext_vector_type(16))) _Float16 v16h;
typedef __attribute__((ext_vector_type(8)))  float    v8f;

#define HCH 128  // hidden channels

// ---------------- aggregation ----------------

__global__ void gin_init_agg(const float* __restrict__ h, const float* __restrict__ eps,
                             int l, float* __restrict__ agg, long long n) {
    long long i = (long long)blockIdx.x * blockDim.x + threadIdx.x;
    if (i < n) agg[i] = (1.0f + eps[l]) * h[i];
}

// one wave (32 lanes) per edge; wave-uniform edge id so src/dst become s_loads;
// 4 channels per lane (float4 gather + 4 atomics into L2-resident agg)
__global__ void gin_scatter_edges(const float* __restrict__ h,
                                  const long long* __restrict__ ei, int E,
                                  float* __restrict__ agg) {
    const int wave = threadIdx.x >> 5;           // wave-uniform
    const int lane = threadIdx.x & 31;
    const long long edge = (long long)blockIdx.x * (blockDim.x >> 5) + wave;
    if (edge >= E) return;
    long long src = ei[edge];                    // edge_index[0][e]  (scalar load)
    long long dst = ei[(long long)E + edge];     // edge_index[1][e]  (scalar load)
    float4 v = ((const float4*)(h + src * HCH))[lane];
    float* a = agg + dst * HCH + lane * 4;
    atomicAdd(a + 0, v.x);
    atomicAdd(a + 1, v.y);
    atomicAdd(a + 2, v.z);
    atomicAdd(a + 3, v.w);
}

// ---------------- WMMA GEMM ----------------
// C[N x M] = act(A[N x K]) * W[K x M] + bias;  act = optional per-column BN+ReLU on A.
// LDS weight swizzle: 16-aligned K-blocks rotated by column so adjacent columns hit
// different banks; each B-fragment run of 16 halfs stays contiguous (ds_load_b128 x2).
__device__ __forceinline__ int widx(int col, int k, int K) {
    int kb = k & ~15;
    int r  = ((kb + (col << 4)) & (K - 1)) | (k & 15);
    return col * K + r;
}

template <int K, int M, bool ACT>
__global__ void gin_gemm_wmma(const float* __restrict__ A, const float* __restrict__ W,
                              const float* __restrict__ bias,
                              const float* __restrict__ scale, const float* __restrict__ shift,
                              float* __restrict__ C, int N) {
    __shared__ _Float16 ldsW[K * M];

    // stage weights f32 -> f16 into LDS (swizzled)
    for (int i = threadIdx.x; i < K * M; i += blockDim.x) {
        int k = i / M, m = i % M;
        ldsW[widx(m, k, K)] = (_Float16)W[i];
    }
    __syncthreads();

    const int wave = threadIdx.x >> 5;
    const int lane = threadIdx.x & 31;
    const int row0 = (blockIdx.x * (blockDim.x >> 5) + wave) * 16;
    if (row0 >= N) return;  // wave-uniform: EXEC stays all-ones for WMMA

    const int hh  = lane >> 4;    // lane half
    const int row = row0 + (lane & 15);

    // preload all A fragments for this 16-row strip (contiguous layout
    // matching B: element e in [0..15] -> K = h*16+e)
    constexpr int NK = K / 32;
    v16h afr[NK];
#pragma unroll
    for (int kk = 0; kk < NK; ++kk) {
        const float* base = A + (size_t)row * K + kk * 32 + hh * 16;
        v16h a;
#pragma unroll
        for (int g = 0; g < 2; ++g) {
            const float* p = base + g * 8;
            int kbase = kk * 32 + hh * 16 + g * 8;
#pragma unroll
            for (int j = 0; j < 8; ++j) {
                float xv = p[j];
                if constexpr (ACT) {
                    xv = fmaxf(scale[kbase + j] * xv + shift[kbase + j], 0.0f);
                }
                a[g * 8 + j] = (_Float16)xv;
            }
        }
        afr[kk] = a;
    }

    const int colb = lane & 15;
    float* crow = C + (size_t)(row0 + 8 * hh) * M + colb;  // one base per lane
#pragma unroll
    for (int col0 = 0; col0 < M; col0 += 16) {
        v8f acc = {};
#pragma unroll
        for (int kk = 0; kk < NK; ++kk) {
            // B fragment (contiguous order, same as A: element e -> K = h*16+e of column col)
            const _Float16* p = &ldsW[widx(col0 + colb, kk * 32 + hh * 16, K)];
            v16h b;
#pragma unroll
            for (int e = 0; e < 16; ++e) b[e] = p[e];
            acc = __builtin_amdgcn_wmma_f32_16x16x32_f16(
                false, afr[kk], false, b, (short)0, acc, false, false);
        }
        float bv = bias[col0 + colb];
#pragma unroll
        for (int v = 0; v < 8; ++v)  // D row = v + 8*h ; offsets fold to immediates
            crow[(size_t)v * M + col0] = acc[v] + bv;
    }
}

// ---------------- batch-norm (training-mode batch stats) ----------------

__global__ void gin_zero(float* __restrict__ p, int n) {
    int i = blockIdx.x * blockDim.x + threadIdx.x;
    if (i < n) p[i] = 0.0f;
}

__global__ void gin_bn_stats(const float* __restrict__ Z, int n, int m,
                             float* __restrict__ sums) {
    int c  = threadIdx.x % m;
    int rl = threadIdx.x / m;
    int tpr = blockDim.x / m;
    long long rows_per_block = (n + gridDim.x - 1) / gridDim.x;
    long long r0 = (long long)blockIdx.x * rows_per_block;
    long long r1 = r0 + rows_per_block; if (r1 > n) r1 = n;
    float s = 0.0f, q = 0.0f;
    for (long long r = r0 + rl; r < r1; r += tpr) {
        float v = Z[r * m + c];
        s += v; q += v * v;
    }
    atomicAdd(&sums[c], s);
    atomicAdd(&sums[m + c], q);
}

__global__ void gin_bn_finalize(const float* __restrict__ sums,
                                const float* __restrict__ g, const float* __restrict__ b,
                                int m, float invN,
                                float* __restrict__ scale, float* __restrict__ shift) {
    int c = threadIdx.x;
    if (c >= m) return;
    float mean = sums[c] * invN;
    float var  = sums[m + c] * invN - mean * mean;
    float r    = rsqrtf(var + 1e-5f);
    float sc   = g[c] * r;
    scale[c] = sc;
    shift[c] = b[c] - mean * sc;
}

__global__ void gin_bn_apply_relu(const float* __restrict__ z,
                                  const float* __restrict__ scale,
                                  const float* __restrict__ shift,
                                  float* __restrict__ out, long long n, int m) {
    long long i = (long long)blockIdx.x * blockDim.x + threadIdx.x;
    if (i >= n) return;
    int c = (int)(i % m);
    out[i] = fmaxf(scale[c] * z[i] + shift[c], 0.0f);
}

// ---------------- log-softmax (in place, per row) ----------------

__global__ void gin_log_softmax(float* __restrict__ Z, int n, int m) {
    int r = blockIdx.x * blockDim.x + threadIdx.x;
    if (r >= n) return;
    float* row = Z + (size_t)r * m;
    float mx = -3.402823466e38f;
    for (int c = 0; c < m; ++c) mx = fmaxf(mx, row[c]);
    float s = 0.0f;
    for (int c = 0; c < m; ++c) s += __expf(row[c] - mx);
    float lse = mx + __logf(s);
    for (int c = 0; c < m; ++c) row[c] = row[c] - lse;
}

// ---------------- host orchestration ----------------

static inline void run_bn(const float* Z, int n, int m, const float* g, const float* b,
                          float* sums, float* scale, float* shift, hipStream_t s) {
    gin_zero<<<1, 512, 0, s>>>(sums, 2 * m);
    gin_bn_stats<<<512, 256, 0, s>>>(Z, n, m, sums);
    gin_bn_finalize<<<1, 256, 0, s>>>(sums, g, b, m, 1.0f / n, scale, shift);
}

extern "C" void kernel_launch(void* const* d_in, const int* in_sizes, int n_in,
                              void* d_out, int out_size, void* d_ws, size_t ws_size,
                              hipStream_t stream) {
    const float*     x   = (const float*)d_in[0];
    const long long* ei  = (const long long*)d_in[1];
    const float*     W1  = (const float*)d_in[2];
    const float*     b1  = (const float*)d_in[3];
    const float*     g1  = (const float*)d_in[4];
    const float*     be1 = (const float*)d_in[5];
    const float*     W2  = (const float*)d_in[6];
    const float*     b2  = (const float*)d_in[7];
    const float*     eps = (const float*)d_in[8];
    const float*     g2  = (const float*)d_in[9];
    const float*     be2 = (const float*)d_in[10];
    const float*     Wl1 = (const float*)d_in[11];
    const float*     bl1 = (const float*)d_in[12];
    const float*     gl  = (const float*)d_in[13];
    const float*     bel = (const float*)d_in[14];
    const float*     Wl2 = (const float*)d_in[15];
    const float*     bl2 = (const float*)d_in[16];

    const int H = HCH, OUT = 64, L = 3;
    const int N = in_sizes[0] / H;
    const int E = in_sizes[1] / 2;

    // workspace layout (floats)
    float* ws    = (float*)d_ws;
    float* agg   = ws;                          // N*H   (reused as z2)
    float* z1    = agg  + (size_t)N * H;        // N*2H  (reused as head z)
    float* hbuf  = z1   + (size_t)N * 2 * H;    // N*H
    float* sums  = hbuf + (size_t)N * H;        // 512
    float* scale = sums + 512;                  // 256
    float* shift = scale + 256;                 // 256

    const dim3 gblk((N + 127) / 128);           // 8 waves/block, 16 rows/wave
    const int  tpb = 256;
    const int  nElemBlocks = (N * H + tpb - 1) / tpb;
    const int  nEdgeBlocks = (int)(((long long)E * 8 + (tpb >> 5) - 1) / (tpb >> 5) / 8 + 1);
    const int  edgeBlocks  = (E + (tpb >> 5) - 1) / (tpb >> 5);  // one wave per edge
    (void)nEdgeBlocks;

    const float* hcur = x;
    for (int l = 0; l < L; ++l) {
        // agg = (1+eps)*h + segment_sum(h[src] -> dst)
        gin_init_agg<<<nElemBlocks, tpb, 0, stream>>>(hcur, eps, l, agg, (long long)N * H);
        gin_scatter_edges<<<edgeBlocks, tpb, 0, stream>>>(hcur, ei, E, agg);

        // z1 = agg @ W1[l] + b1[l]        (N x 2H), K=128
        gin_gemm_wmma<128, 256, false><<<gblk, tpb, 0, stream>>>(
            agg, W1 + (size_t)l * H * 2 * H, b1 + (size_t)l * 2 * H,
            nullptr, nullptr, z1, N);

        // BN(2H) stats; then z2 = relu(bn(z1)) @ W2[l] + b2[l]  (fused into A-load), K=256
        run_bn(z1, N, 2 * H, g1 + (size_t)l * 2 * H, be1 + (size_t)l * 2 * H,
               sums, scale, shift, stream);
        gin_gemm_wmma<256, 128, true><<<gblk, tpb, 0, stream>>>(
            z1, W2 + (size_t)l * 2 * H * H, b2 + (size_t)l * H,
            scale, shift, agg, N);

        // h = relu(bn(z2))  (outer BN)
        run_bn(agg, N, H, g2 + (size_t)l * H, be2 + (size_t)l * H,
               sums, scale, shift, stream);
        gin_bn_apply_relu<<<nElemBlocks, tpb, 0, stream>>>(agg, scale, shift, hbuf,
                                                           (long long)N * H, H);
        hcur = hbuf;
    }

    // head: z = h @ Wl1 + bl1 ; relu(bn(z)) @ Wl2 + bl2 ; log_softmax
    gin_gemm_wmma<128, 128, false><<<gblk, tpb, 0, stream>>>(
        hcur, Wl1, bl1, nullptr, nullptr, z1, N);
    run_bn(z1, N, H, gl, bel, sums, scale, shift, stream);
    gin_gemm_wmma<128, 64, true><<<gblk, tpb, 0, stream>>>(
        z1, Wl2, bl2, scale, shift, (float*)d_out, N);
    gin_log_softmax<<<(N + tpb - 1) / tpb, tpb, 0, stream>>>((float*)d_out, N, OUT);
}
